// MP_Layer_25958782337708
// MI455X (gfx1250) — matrix operand order repaired and checked by reading, not compiled, not run on hardware
//
#include <hip/hip_runtime.h>

typedef __bf16 bf16;
typedef __attribute__((ext_vector_type(16))) __bf16 v16bf;
typedef __attribute__((ext_vector_type(8)))  __bf16 v8bf;
typedef __attribute__((ext_vector_type(8)))  float  v8f;

#define LOG2E 1.4426950408889634f

static __device__ __forceinline__ v16bf cat16(v8bf lo, v8bf hi) {
  return __builtin_shufflevector(lo, hi, 0,1,2,3,4,5,6,7,8,9,10,11,12,13,14,15);
}
static __device__ __forceinline__ v8f wmma_bf16(v16bf a, v16bf b, v8f c) {
  // 8 args: (neg_a, A, neg_b, B, c_mod, C, reuse_a, reuse_b)
  return __builtin_amdgcn_wmma_f32_16x16x32_bf16(false, a, false, b, (short)0, c, false, false);
}

// ---------------------------------------------------------------------------
// Kernel 0: pack / precision conversion.
//  - h -> bf16 [8192][128]
//  - coords -> Crow[i]=(sx,sy,sz,q), Pcol[j]=(-2sx,-2sy,-2sz,q), log2e-scaled
//    so dist2_scaled(i,j) = Crow.w + Pcol.w + dot(Crow.xyz, Pcol.xyz)
//  - W_self/W_msg/W_upd -> bf16 copies
// ---------------------------------------------------------------------------
__global__ __launch_bounds__(256) void pack_kernel(
    const float* __restrict__ h, const float* __restrict__ coords,
    const float* __restrict__ Wself, const float* __restrict__ Wmsg,
    const float* __restrict__ Wupd,
    bf16* __restrict__ hb, bf16* __restrict__ wsb, bf16* __restrict__ wmb,
    bf16* __restrict__ wub, float4* __restrict__ Crow, float4* __restrict__ Pcol) {
  int idx = blockIdx.x * 256 + threadIdx.x;          // 0 .. 1048575
  hb[idx] = (bf16)h[idx];
  if (idx < 8192) {
    float x = coords[3*idx+0] * LOG2E;
    float y = coords[3*idx+1] * LOG2E;
    float z = coords[3*idx+2] * LOG2E;
    float q = x*x + y*y + z*z;
    Crow[idx] = make_float4(x, y, z, q);
    Pcol[idx] = make_float4(-2.0f*x, -2.0f*y, -2.0f*z, q);
  }
  if (idx < 16384) { wsb[idx] = (bf16)Wself[idx]; wmb[idx] = (bf16)Wmsg[idx]; }
  if (idx < 32768) { wub[idx] = (bf16)Wupd[idx]; }
}

// ---------------------------------------------------------------------------
// Kernel 1: msg_in = h @ Wmsg^T + b_msg (store bf16 TRANSPOSED [B][128][2048],
//           with mask_j folded in), h_self = h @ Wself^T + b_self (bf16 row-major).
// One wave -> 16 rows x 16 output cols, K=128 (4 WMMAs per matrix).
// ---------------------------------------------------------------------------
__global__ __launch_bounds__(256) void gemm1_kernel(
    const bf16* __restrict__ hb, const bf16* __restrict__ wsb,
    const bf16* __restrict__ wmb, const float* __restrict__ bself,
    const float* __restrict__ bmsg, const float* __restrict__ mask,
    bf16* __restrict__ hself, bf16* __restrict__ msginT) {
  int wid  = (blockIdx.x * blockDim.x + threadIdx.x) >> 5;   // 0..4095
  int lane = threadIdx.x & 31;
  int rb = wid >> 3;                  // row block over 8192 rows
  int d0 = (wid & 7) * 16;            // output column tile
  int r = lane & 15, hi = lane >> 4, haloff = hi * 8;
  int row = rb * 16 + r;

  const bf16* arow  = hb  + (size_t)row * 128;
  const bf16* wsrow = wsb + (size_t)(d0 + r) * 128;   // lane n == r holds column d0+n
  const bf16* wmrow = wmb + (size_t)(d0 + r) * 128;

  v8f accS = {}; v8f accM = {};
#pragma unroll
  for (int kt = 0; kt < 4; ++kt) {
    int k0 = kt * 32;
    v16bf a = cat16(*(const v8bf*)(arow + k0 + haloff),
                    *(const v8bf*)(arow + k0 + 16 + haloff));
    v16bf bs = cat16(*(const v8bf*)(wsrow + k0 + haloff),
                     *(const v8bf*)(wsrow + k0 + 16 + haloff));
    accS = wmma_bf16(a, bs, accS);
    v16bf bm = cat16(*(const v8bf*)(wmrow + k0 + haloff),
                     *(const v8bf*)(wmrow + k0 + 16 + haloff));
    accM = wmma_bf16(a, bm, accM);
  }

  float bsv = bself[d0 + r], bmv = bmsg[d0 + r];
  int b = rb >> 7;                    // batch (128 row-blocks per batch)
  int jbase = rb * 16 + hi * 8;       // this lane holds rows jbase..jbase+7
  int jloc  = jbase & 2047;
#pragma unroll
  for (int v = 0; v < 8; ++v) {
    int j = jbase + v;
    hself[(size_t)j * 128 + d0 + r] = (bf16)(accS[v] + bsv);
    float mg = (accM[v] + bmv) * mask[j];          // fold mask_j
    msginT[((size_t)b * 128 + d0 + r) * 2048 + jloc + v] = (bf16)mg;
  }
}

// ---------------------------------------------------------------------------
// Kernel 2 (hot): msg[i,:] = sum_j exp(-dist(i,j)) * msg_in_masked[j,:]
// Workgroup-cooperative K-split: 8 waves per block, one 16x128 output slab per
// block; wave w owns K range [w*256, w*256+256) (8 WMMA steps of 32).
// W tiles generated per step directly in the WMMA A-register layout; partial
// f32 accumulators reduced through LDS, mask_i applied, stored bf16.
// ---------------------------------------------------------------------------
__global__ __launch_bounds__(256) void msg_kernel(
    const float4* __restrict__ Crow, const float4* __restrict__ Pcol,
    const bf16* __restrict__ msginT, const float* __restrict__ mask,
    bf16* __restrict__ msg) {
  __shared__ float red[8 * 2048];   // 64 KB: [wave][tile][lane][v]

  int tid  = threadIdx.x;
  int w    = tid >> 5;              // wave 0..7 -> K slice
  int lane = tid & 31;
  int blk  = blockIdx.x;            // 0..511
  int b  = blk >> 7;                // 128 blocks per batch
  int i0 = (blk & 127) * 16;
  int r = lane & 15, hi = lane >> 4, haloff = hi * 8;

  float4 cr = Crow[b * 2048 + i0 + r];
  const float4* pc = Pcol + b * 2048;
  const bf16* bT = msginT + (size_t)b * 128 * 2048;

  v8f acc[8];
#pragma unroll
  for (int t = 0; t < 8; ++t) acc[t] = (v8f){0,0,0,0,0,0,0,0};

  int jbeg = w * 256;
  for (int j0 = jbeg; j0 < jbeg + 256; j0 += 32) {
    v16bf a;
#pragma unroll
    for (int t = 0; t < 8; ++t) {
      float4 p = pc[j0 + haloff + t];
      float d2 = cr.w + p.w;
      d2 = fmaf(cr.x, p.x, d2);
      d2 = fmaf(cr.y, p.y, d2);
      d2 = fmaf(cr.z, p.z, d2);
      d2 = fmaxf(d2, 0.0f);
      a[t] = (bf16)__builtin_amdgcn_exp2f(-__builtin_amdgcn_sqrtf(d2));
      float4 p2 = pc[j0 + 16 + haloff + t];
      float e2 = cr.w + p2.w;
      e2 = fmaf(cr.x, p2.x, e2);
      e2 = fmaf(cr.y, p2.y, e2);
      e2 = fmaf(cr.z, p2.z, e2);
      e2 = fmaxf(e2, 0.0f);
      a[t + 8] = (bf16)__builtin_amdgcn_exp2f(-__builtin_amdgcn_sqrtf(e2));
    }
#pragma unroll
    for (int t = 0; t < 8; ++t) {
      const bf16* bp = bT + (size_t)(t * 16 + r) * 2048 + j0 + haloff;
      acc[t] = wmma_bf16(a, cat16(*(const v8bf*)bp, *(const v8bf*)(bp + 16)), acc[t]);
    }
  }

  // dump partial accumulators to LDS: addr = ((w*8 + t)*32 + lane)*8 + v
#pragma unroll
  for (int t = 0; t < 8; ++t)
    *(v8f*)&red[((w * 8 + t) * 32 + lane) * 8] = acc[t];
  __syncthreads();

  // cross-wave reduction: each thread owns 8 output elements (coalesced in d)
#pragma unroll
  for (int e8 = 0; e8 < 8; ++e8) {
    int e = tid + e8 * 256;         // 0..2047 over (m, d)
    int d = e & 127;
    int m = e >> 7;                 // 0..15
    int t = d >> 4, n = d & 15;
    int li = n + ((m & 8) ? 16 : 0);
    int v = m & 7;
    float s = 0.0f;
#pragma unroll
    for (int ww = 0; ww < 8; ++ww)
      s += red[((ww * 8 + t) * 32 + li) * 8 + v];
    int irow = b * 2048 + i0 + m;
    msg[(size_t)irow * 128 + d] = (bf16)(s * mask[irow]);   // fold mask_i
  }
}

// ---------------------------------------------------------------------------
// Kernel 3: out = [h_self | msg] @ W_upd^T + b_upd   (K=256, f32 output)
// ---------------------------------------------------------------------------
__global__ __launch_bounds__(256) void final_kernel(
    const bf16* __restrict__ hself, const bf16* __restrict__ msg,
    const bf16* __restrict__ wub, const float* __restrict__ bupd,
    float* __restrict__ out) {
  int wid  = (blockIdx.x * blockDim.x + threadIdx.x) >> 5;   // 0..4095
  int lane = threadIdx.x & 31;
  int rb = wid >> 3;
  int d0 = (wid & 7) * 16;
  int r = lane & 15, hi = lane >> 4, haloff = hi * 8;
  int row = rb * 16 + r;

  const bf16* a1 = hself + (size_t)row * 128;
  const bf16* a2 = msg   + (size_t)row * 128;
  const bf16* wrow = wub + (size_t)(d0 + r) * 256;

  v8f acc = {};
#pragma unroll
  for (int kt = 0; kt < 8; ++kt) {
    int k0 = kt * 32;
    const bf16* asrc = (k0 < 128) ? (a1 + k0) : (a2 + (k0 - 128));
    v16bf a = cat16(*(const v8bf*)(asrc + haloff),
                    *(const v8bf*)(asrc + 16 + haloff));
    v16bf bm = cat16(*(const v8bf*)(wrow + k0 + haloff),
                     *(const v8bf*)(wrow + k0 + 16 + haloff));
    acc = wmma_bf16(a, bm, acc);
  }
  float bias = bupd[d0 + r];
  int obase = rb * 16 + hi * 8;
#pragma unroll
  for (int v = 0; v < 8; ++v)
    out[(size_t)(obase + v) * 128 + d0 + r] = acc[v] + bias;
}

// ---------------------------------------------------------------------------
extern "C" void kernel_launch(void* const* d_in, const int* in_sizes, int n_in,
                              void* d_out, int out_size, void* d_ws, size_t ws_size,
                              hipStream_t stream) {
  const float* h      = (const float*)d_in[0];
  const float* coords = (const float*)d_in[1];
  const float* mask   = (const float*)d_in[2];
  const float* Wself  = (const float*)d_in[3];
  const float* bself  = (const float*)d_in[4];
  const float* Wmsg   = (const float*)d_in[5];
  const float* bmsg   = (const float*)d_in[6];
  const float* Wupd   = (const float*)d_in[7];
  const float* bupd   = (const float*)d_in[8];

  char* ws = (char*)d_ws;
  bf16*   hb     = (bf16*)  (ws + 0);          // 2 MB
  bf16*   wsb    = (bf16*)  (ws + 2097152);    // 32 KB
  bf16*   wmb    = (bf16*)  (ws + 2129920);    // 32 KB
  bf16*   wub    = (bf16*)  (ws + 2162688);    // 64 KB
  float4* Crow   = (float4*)(ws + 2228224);    // 128 KB
  float4* Pcol   = (float4*)(ws + 2359296);    // 128 KB
  bf16*   msginT = (bf16*)  (ws + 2490368);    // 2 MB
  bf16*   hself  = (bf16*)  (ws + 4587520);    // 2 MB
  bf16*   msgb   = (bf16*)  (ws + 6684672);    // 2 MB

  pack_kernel <<<4096, 256, 0, stream>>>(h, coords, Wself, Wmsg, Wupd,
                                         hb, wsb, wmb, wub, Crow, Pcol);
  gemm1_kernel<<<512, 256, 0, stream>>>(hb, wsb, wmb, bself, bmsg, mask,
                                        hself, msginT);
  msg_kernel  <<<512, 256, 0, stream>>>(Crow, Pcol, msginT, mask, msgb);
  final_kernel<<<512, 256, 0, stream>>>(hself, msgb, wub, bupd, (float*)d_out);
}
